// MultiheadAttentionWithAttention_12816182411755
// MI455X (gfx1250) — compile-verified, matrix operand order checked by reading, not run on hardware
//
#include <hip/hip_runtime.h>
#include <hip/hip_bf16.h>

typedef _Float16 v16h __attribute__((ext_vector_type(16)));
typedef _Float16 v4h  __attribute__((ext_vector_type(4)));
typedef float    v8f  __attribute__((ext_vector_type(8)));

union V16H { v16h v; uint4 u[2]; };
union V8F  { v8f  v; float f[8]; };

// ---------------------------------------------------------------------------
// Fragment loader for the CDNA5 16-bit 16x32 WMMA operand layout.
// lane r = lane&15 selects the row (A) / column (B); hi = lane>>4 selects the
// K half: halves 0..7 <- K = k0 + hi*8 + j ; halves 8..15 <- K = k0+16+hi*8+j.
// p must point at the start of the (r-th) row; rows are 16B aligned.
// ---------------------------------------------------------------------------
__device__ __forceinline__ v16h ldfrag(const _Float16* p, int k0, int hi) {
  V16H x;
  x.u[0] = *(const uint4*)(p + k0 + hi * 8);
  x.u[1] = *(const uint4*)(p + k0 + 16 + hi * 8);
  return x.v;
}

__device__ __forceinline__ v8f wmma_f16(v16h a, v16h b, v8f c) {
  return __builtin_amdgcn_wmma_f32_16x16x32_f16(false, a, false, b,
                                                (short)0, c, false, false);
}

// ---------------------------------------------------------------------------
// f32 -> f16 conversion (x4 vectorized)
// ---------------------------------------------------------------------------
__global__ void cvt_f32_f16_x4(const float4* __restrict__ in,
                               v4h* __restrict__ out, int n4) {
  int i = blockIdx.x * blockDim.x + threadIdx.x;
  int st = gridDim.x * blockDim.x;
  for (; i < n4; i += st) {
    float4 f = in[i];
    v4h h;
    h[0] = (_Float16)f.x; h[1] = (_Float16)f.y;
    h[2] = (_Float16)f.z; h[3] = (_Float16)f.w;
    out[i] = h;
  }
}

__global__ void zero_f32(float* __restrict__ p, int n) {
  int i = blockIdx.x * blockDim.x + threadIdx.x;
  int st = gridDim.x * blockDim.x;
  for (; i < n; i += st) p[i] = 0.0f;
}

// ---------------------------------------------------------------------------
// GEMM: out[m,n] = sum_k A[m,k] * W[n,k] + bias[n]
// A: [M,1024] f16 row-major, W: [1024,1024] f16 row-major (torch Linear).
// Both operands are K-contiguous -> identical fragment loads.
// MODE 0: f32 out [M,1024]
// MODE 1: f16 out, head-split  [B,H,T,64]
// MODE 2: f16 out, head-transposed [B,H,64,T_MEM]   (for V)
// Block tile 128x64, 8 waves (wave32) in a 4x2 grid, 32x32 per wave.
// ---------------------------------------------------------------------------
template <int MODE>
__global__ __launch_bounds__(256)
void gemm_xwT(const _Float16* __restrict__ A, const _Float16* __restrict__ W,
              const float* __restrict__ bias, void* __restrict__ outp, int M) {
  __shared__ _Float16 As[128][72];   // 144B rows, 16B aligned, bank-padded
  __shared__ _Float16 Bs[64][72];

  const int tid = threadIdx.x;
  const int n0 = blockIdx.x * 64;
  const int m0 = blockIdx.y * 128;
  const int wv = tid >> 5, lane = tid & 31;
  const int r = lane & 15, hi = lane >> 4;
  const int wm = (wv >> 1) * 32;     // wave M offset in block tile
  const int wn = (wv & 1) * 32;      // wave N offset in block tile

  V8F c[2][2];
#pragma unroll
  for (int i = 0; i < 2; ++i)
#pragma unroll
    for (int j = 0; j < 2; ++j)
#pragma unroll
      for (int e = 0; e < 8; ++e) c[i][j].f[e] = 0.0f;

  for (int kb = 0; kb < 1024; kb += 64) {
    // stage A 128x64 (4 x b128 per thread) and W 64x64 (2 x b128 per thread)
#pragma unroll
    for (int i = 0; i < 4; ++i) {
      int v = tid + 256 * i;
      int row = v >> 3, c8 = (v & 7) * 8;
      *(uint4*)&As[row][c8] =
          *(const uint4*)(A + (size_t)(m0 + row) * 1024 + kb + c8);
    }
#pragma unroll
    for (int i = 0; i < 2; ++i) {
      int v = tid + 256 * i;
      int row = v >> 3, c8 = (v & 7) * 8;
      *(uint4*)&Bs[row][c8] =
          *(const uint4*)(W + (size_t)(n0 + row) * 1024 + kb + c8);
    }
    if (kb + 64 < 1024) {   // global_prefetch_b8 of next K slab
      __builtin_prefetch(A + (size_t)(m0 + (tid >> 1)) * 1024 + kb + 64, 0, 0);
      __builtin_prefetch(W + (size_t)(n0 + (tid & 63)) * 1024 + kb + 64, 0, 0);
    }
    __syncthreads();

#pragma unroll
    for (int kk = 0; kk < 64; kk += 32) {
      v16h a0 = ldfrag(&As[wm + r][0], kk, hi);
      v16h a1 = ldfrag(&As[wm + 16 + r][0], kk, hi);
      v16h b0 = ldfrag(&Bs[wn + r][0], kk, hi);
      v16h b1 = ldfrag(&Bs[wn + 16 + r][0], kk, hi);
      c[0][0].v = wmma_f16(a0, b0, c[0][0].v);
      c[0][1].v = wmma_f16(a0, b1, c[0][1].v);
      c[1][0].v = wmma_f16(a1, b0, c[1][0].v);
      c[1][1].v = wmma_f16(a1, b1, c[1][1].v);
    }
    __syncthreads();
  }

  // store: C element e on lane (r,hi) is (m_local = e + 8*hi, n_local = r)
#pragma unroll
  for (int mt = 0; mt < 2; ++mt)
#pragma unroll
    for (int nt = 0; nt < 2; ++nt) {
      int n = n0 + wn + nt * 16 + r;
      float bn = bias[n];
#pragma unroll
      for (int e = 0; e < 8; ++e) {
        int m = m0 + wm + mt * 16 + e + 8 * hi;
        float val = c[mt][nt].f[e] + bn;
        if (MODE == 0) {
          ((float*)outp)[(size_t)m * 1024 + n] = val;
        } else {
          int b = m >> 10, t = m & 1023, h = n >> 6, d = n & 63;
          _Float16 hv = (_Float16)val;
          if (MODE == 1)
            ((_Float16*)outp)[(((size_t)(b * 16 + h)) * 1024 + t) * 64 + d] = hv;
          else
            ((_Float16*)outp)[(((size_t)(b * 16 + h)) * 64 + d) * 1024 + t] = hv;
        }
      }
    }
}

// ---------------------------------------------------------------------------
// Attention: one workgroup per (b, h, 32-row q block).
// Full 32x1024 score strip in LDS (CDNA5: 320KB/WGP), softmax in place,
// probabilities accumulated (x 1/16) into the mean-attention output, then
// P @ V with WMMA (P converted f32->f16 from LDS, V pre-transposed f16).
// ---------------------------------------------------------------------------
__global__ __launch_bounds__(256)
void attn_kernel(const _Float16* __restrict__ Qh, const _Float16* __restrict__ Kh,
                 const _Float16* __restrict__ Vt, _Float16* __restrict__ Yh,
                 float* __restrict__ attOut) {
  extern __shared__ char smem[];
  float*    S   = (float*)smem;                                   // [32][1032]
  _Float16* Qt  = (_Float16*)(smem + 32 * 1032 * 4);              // [32][72]
  float*    red = (float*)(smem + 32 * 1032 * 4 + 32 * 72 * 2);   // [32][8]

  const int tid  = threadIdx.x;
  const int qblk = blockIdx.x & 31;
  const int h    = (blockIdx.x >> 5) & 15;
  const int b    = blockIdx.x >> 9;
  const int q0   = qblk * 32;
  const int wv = tid >> 5, lane = tid & 31;
  const int r = lane & 15, hi = lane >> 4;

  const _Float16* Qbase = Qh + (((size_t)(b * 16 + h)) * 1024 + q0) * 64;
  const _Float16* Kbase = Kh + ((size_t)(b * 16 + h)) * 1024 * 64;
  const _Float16* Vbase = Vt + ((size_t)(b * 16 + h)) * 64 * 1024;

  { // stage Q tile 32x64 -> LDS (one b128 per thread)
    int row = tid >> 3, c8 = (tid & 7) * 8;
    *(uint4*)&Qt[row * 72 + c8] = *(const uint4*)(Qbase + row * 64 + c8);
  }
  __syncthreads();

  // -------- S = (Q K^T) * 1/sqrt(64), K fragments straight from global ----
  {
    v16h a00 = ldfrag(Qt + (0 + r) * 72, 0, hi);
    v16h a01 = ldfrag(Qt + (0 + r) * 72, 32, hi);
    v16h a10 = ldfrag(Qt + (16 + r) * 72, 0, hi);
    v16h a11 = ldfrag(Qt + (16 + r) * 72, 32, hi);
    for (int nt = wv; nt < 64; nt += 8) {
      const _Float16* kp = Kbase + (size_t)(nt * 16 + r) * 64;
      v16h b0 = ldfrag(kp, 0, hi);
      v16h b1 = ldfrag(kp, 32, hi);
      V8F c0, c1;
#pragma unroll
      for (int e = 0; e < 8; ++e) { c0.f[e] = 0.0f; c1.f[e] = 0.0f; }
      c0.v = wmma_f16(a00, b0, c0.v);
      c0.v = wmma_f16(a01, b1, c0.v);
      c1.v = wmma_f16(a10, b0, c1.v);
      c1.v = wmma_f16(a11, b1, c1.v);
      int col = nt * 16 + r;
#pragma unroll
      for (int e = 0; e < 8; ++e) {
        S[(e + 8 * hi) * 1032 + col]      = c0.f[e] * 0.125f;
        S[(16 + e + 8 * hi) * 1032 + col] = c1.f[e] * 0.125f;
      }
    }
  }
  __syncthreads();

  // -------- softmax over each of the 32 rows (8 threads / row) ------------
  {
    int row = tid >> 3, seg = tid & 7;
    float* Srow = S + row * 1032;
    float mx = -1e30f;
    for (int ci = seg * 128; ci < seg * 128 + 128; ++ci)
      mx = fmaxf(mx, Srow[ci]);
    red[row * 8 + seg] = mx;
    __syncthreads();
    float rm = red[row * 8];
#pragma unroll
    for (int j = 1; j < 8; ++j) rm = fmaxf(rm, red[row * 8 + j]);
    __syncthreads();
    float sm = 0.0f;
    for (int ci = seg * 128; ci < seg * 128 + 128; ++ci) {
      float e = __expf(Srow[ci] - rm);
      Srow[ci] = e;
      sm += e;
    }
    red[row * 8 + seg] = sm;
    __syncthreads();
    float rs = 0.0f;
#pragma unroll
    for (int j = 0; j < 8; ++j) rs += red[row * 8 + j];
    float inv = 1.0f / rs;
    float* arow = attOut + ((size_t)(b * 1024) + q0 + row) * 1024;
    for (int ci = seg * 128; ci < seg * 128 + 128; ++ci) {
      float p = Srow[ci] * inv;
      Srow[ci] = p;
      atomicAdd(&arow[ci], p * 0.0625f);  // mean over 16 heads
    }
  }
  __syncthreads();

  // -------- Y = P @ V : wave -> (mt, nt) tile of the 32x64 output ---------
  {
    int mt = wv >> 2, nt = wv & 3;
    V8F c;
#pragma unroll
    for (int e = 0; e < 8; ++e) c.f[e] = 0.0f;
    const _Float16* vp = Vbase + (size_t)(nt * 16 + r) * 1024;
    const float* srow = S + (mt * 16 + r) * 1032;
    for (int k0 = 0; k0 < 1024; k0 += 32) {
      v16h a;
#pragma unroll
      for (int j = 0; j < 8; ++j) a[j] = (_Float16)srow[k0 + hi * 8 + j];
#pragma unroll
      for (int j = 0; j < 8; ++j) a[8 + j] = (_Float16)srow[k0 + 16 + hi * 8 + j];
      v16h bf = ldfrag(vp, k0, hi);
      c.v = wmma_f16(a, bf, c.v);
    }
#pragma unroll
    for (int e = 0; e < 8; ++e) {
      int qrow = q0 + mt * 16 + e + 8 * hi;
      int d = nt * 16 + r;
      Yh[((size_t)(b * 1024) + qrow) * 1024 + h * 64 + d] = (_Float16)c.f[e];
    }
  }
}

// ---------------------------------------------------------------------------
extern "C" void kernel_launch(void* const* d_in, const int* in_sizes, int n_in,
                              void* d_out, int out_size, void* d_ws,
                              size_t ws_size, hipStream_t stream) {
  const float* q_in = (const float*)d_in[0];
  const float* k_in = (const float*)d_in[1];
  const float* v_in = (const float*)d_in[2];
  const float* Wq = (const float*)d_in[3];
  const float* bq = (const float*)d_in[4];
  const float* Wk = (const float*)d_in[5];
  const float* bk = (const float*)d_in[6];
  const float* Wv = (const float*)d_in[7];
  const float* bv = (const float*)d_in[8];
  const float* Wo = (const float*)d_in[9];
  const float* bo = (const float*)d_in[10];

  constexpr size_t MB = 1024 * 1024;
  char* ws = (char*)d_ws;
  _Float16* q_h  = (_Float16*)(ws + 0 * MB);    // 8MB each (4M halves)
  _Float16* k_h  = (_Float16*)(ws + 8 * MB);
  _Float16* v_h  = (_Float16*)(ws + 16 * MB);
  _Float16* wq_h = (_Float16*)(ws + 24 * MB);   // 2MB each
  _Float16* wk_h = (_Float16*)(ws + 26 * MB);
  _Float16* wv_h = (_Float16*)(ws + 28 * MB);
  _Float16* wo_h = (_Float16*)(ws + 30 * MB);
  _Float16* Qh   = (_Float16*)(ws + 32 * MB);   // [B,H,T,64]
  _Float16* Kh   = (_Float16*)(ws + 40 * MB);   // [B,H,T_MEM,64]
  _Float16* Vt   = (_Float16*)(ws + 48 * MB);   // [B,H,64,T_MEM]
  _Float16* Yh   = (_Float16*)(ws + 56 * MB);   // [B,T,C] pre-projection

  const int ACT4 = (4 * 1024 * 1024) / 4;   // 1M float4 per activation
  const int WT4  = (1024 * 1024) / 4;       // 256K float4 per weight

  cvt_f32_f16_x4<<<2048, 256, 0, stream>>>((const float4*)q_in, (v4h*)q_h, ACT4);
  cvt_f32_f16_x4<<<2048, 256, 0, stream>>>((const float4*)k_in, (v4h*)k_h, ACT4);
  cvt_f32_f16_x4<<<2048, 256, 0, stream>>>((const float4*)v_in, (v4h*)v_h, ACT4);
  cvt_f32_f16_x4<<<1024, 256, 0, stream>>>((const float4*)Wq, (v4h*)wq_h, WT4);
  cvt_f32_f16_x4<<<1024, 256, 0, stream>>>((const float4*)Wk, (v4h*)wk_h, WT4);
  cvt_f32_f16_x4<<<1024, 256, 0, stream>>>((const float4*)Wv, (v4h*)wv_h, WT4);
  cvt_f32_f16_x4<<<1024, 256, 0, stream>>>((const float4*)Wo, (v4h*)wo_h, WT4);

  float* y_out   = (float*)d_out;
  float* att_out = (float*)d_out + (size_t)4 * 1024 * 1024;
  zero_f32<<<2048, 256, 0, stream>>>(att_out, 4 * 1024 * 1024);

  dim3 ggrid(16, 32);  // N/64 x M/128
  gemm_xwT<1><<<ggrid, 256, 0, stream>>>(q_h, wq_h, bq, Qh, 4096);
  gemm_xwT<1><<<ggrid, 256, 0, stream>>>(k_h, wk_h, bk, Kh, 4096);
  gemm_xwT<2><<<ggrid, 256, 0, stream>>>(v_h, wv_h, bv, Vt, 4096);

  // 32*1032 f32 scores + 32*72 f16 Q tile + 32*8 f32 reduction scratch
  const size_t smem = 32 * 1032 * 4 + 32 * 72 * 2 + 32 * 8 * 4;  // 137728 B
  attn_kernel<<<4 * 16 * 32, 256, smem, stream>>>(Qh, Kh, Vt, Yh, att_out);

  gemm_xwT<0><<<ggrid, 256, 0, stream>>>(Yh, wo_h, bo, y_out, 4096);
}
